// TransformerBlock_47562467836364
// MI455X (gfx1250) — compile-verified, matrix operand order checked by reading, not run on hardware
//
#include <hip/hip_runtime.h>
#include <hip/hip_bf16.h>

// ---------------------------------------------------------------- constants
#define BB   2
#define NN   2048
#define DD_  1024
#define HH   16
#define HD_  64
#define EE   8
#define DFF  4096
#define TOK  (BB*NN)          // 4096
#define KCAP (TOK/EE)         // 512

typedef __attribute__((ext_vector_type(16))) _Float16 v16h;
typedef __attribute__((ext_vector_type(8)))  float    v8f;

// ---------------------------------------------------------------- helpers
__device__ __forceinline__ float wave_sum(float v) {
    for (int o = 16; o > 0; o >>= 1) v += __shfl_xor(v, o, 32);
    return v;
}

// CDNA5 async memory->LDS copy (16B per lane), tracked by ASYNCcnt.
// Generic LDS pointer's low 32 bits are the LDS byte offset (aperture layout).
__device__ __forceinline__ void async_copy16(const _Float16* g, _Float16* l) {
    unsigned int  lofs = (unsigned int)(unsigned long long)l;
    unsigned long long ga = (unsigned long long)g;
    asm volatile("global_load_async_to_lds_b128 %0, %1, off"
                 :: "v"(lofs), "v"(ga)
                 : "memory");
}
__device__ __forceinline__ void wait_async0() {
    asm volatile("s_wait_asynccnt 0x0" ::: "memory");
}

// WMMA fragment loader from LDS tile.
// Per ISA layout: element e of a lane maps to k = e + 8*((e>>3) + (lane>>4)),
// so a fragment is two contiguous 16B chunks at kOff + 8*(lane>>4) and +16.
__device__ __forceinline__ v16h ld_frag(const _Float16* tile, int row, int stride,
                                        int kOff, int lane) {
    const _Float16* p = tile + row * stride + kOff + ((lane >> 4) << 3);
    union { uint4 u[2]; v16h v; } f;
    f.u[0] = *(const uint4*)(p);
    f.u[1] = *(const uint4*)(p + 16);
    return f.v;
}

__device__ __forceinline__ v8f wmma16(v16h a, v16h b, v8f c) {
    return __builtin_amdgcn_wmma_f32_16x16x32_f16(
        false, a, false, b, (short)0, c, false, false);
}

__device__ __forceinline__ float gelu_tanh(float x) {
    float x3 = x * x * x;
    return 0.5f * x * (1.0f + tanhf(0.7978845608028654f * (x + 0.044715f * x3)));
}

// ---------------------------------------------------------------- shift GEMV
// shift[b][j] = sum_k silu(t[b][k]) * W_ada[j][k] + b_ada[j]
__global__ __launch_bounds__(256) void shift_kernel(const float* __restrict__ t,
                                                    const float* __restrict__ Wada,
                                                    const float* __restrict__ bada,
                                                    float* __restrict__ shift) {
    int gid = blockIdx.x * 256 + threadIdx.x;       // 0..2047
    int b = gid >> 10, j = gid & 1023;
    const float* tr = t + (long)b * DD_;
    const float* wr = Wada + (long)j * DD_;
    float acc = 0.f;
    for (int k = 0; k < DD_; ++k) {
        float tv = tr[k];
        acc += (tv / (1.f + expf(-tv))) * wr[k];
    }
    shift[gid] = acc + bada[j];
}

// ---------------------------------------------------------------- LN kernel
// One block per row of 1024. selIdx (optional) gathers rows; shift (optional)
// adds per-batch shift. Output f16.
__global__ __launch_bounds__(256) void ln_kernel(const float* __restrict__ x,
                                                 const int* __restrict__ selIdx,
                                                 const float* __restrict__ shift,
                                                 const float* __restrict__ g,
                                                 const float* __restrict__ bvec,
                                                 _Float16* __restrict__ out) {
    __shared__ float rs[8], rs2[8];
    __shared__ float sM, sR;
    int tid = threadIdx.x, lane = tid & 31, wid = tid >> 5;
    long row = blockIdx.x;
    long src = selIdx ? (long)selIdx[row] : row;
    const float* xr = x + src * DD_;
    float4 v = ((const float4*)xr)[tid];
    float s  = v.x + v.y + v.z + v.w;
    float s2 = v.x*v.x + v.y*v.y + v.z*v.z + v.w*v.w;
    s = wave_sum(s); s2 = wave_sum(s2);
    if (lane == 0) { rs[wid] = s; rs2[wid] = s2; }
    __syncthreads();
    if (tid == 0) {
        float a = 0.f, b = 0.f;
        for (int i = 0; i < 8; ++i) { a += rs[i]; b += rs2[i]; }
        float m = a * (1.0f / DD_);
        sM = m;
        sR = rsqrtf(b * (1.0f / DD_) - m * m + 1e-5f);
    }
    __syncthreads();
    float m = sM, r = sR;
    int bidx = (int)(row >> 11);                    // token / 2048 (batch)
    #pragma unroll
    for (int c = 0; c < 4; ++c) {
        int j = tid * 4 + c;
        float xv = (&v.x)[c];
        float o = (xv - m) * r * g[j] + bvec[j];
        if (shift) o += shift[(long)bidx * DD_ + j];
        out[row * DD_ + j] = (_Float16)o;
    }
}

// ---------------------------------------------------------------- generic GEMM
// C(MxN) = A(MxK,f16) * W(NxK,f32)^T ; 128x128x64 tiles, 8 waves (2x4).
// A tile copied via CDNA5 async-to-LDS; W tile converted f32->f16 in-register.
struct GemmArgs {
    const _Float16* A;  long aE;
    const float*    W;  long wE;
    const float*    bias; long bE;
    float*    outF;     long oE;
    _Float16* outH;
    const float* resid;
    const int*   selIdx;
    const float* selW;  long selE;
    float* scatter;
    int M, N, K;
};

#define GM_F16      0
#define GM_RESID    1
#define GM_GELU_F16 2
#define GM_SCATTER  3

template <int MODE>
__global__ __launch_bounds__(256) void gemm_kernel(GemmArgs p) {
    __shared__ _Float16 sA[128 * 72];
    __shared__ _Float16 sB[128 * 72];

    int tid = threadIdx.x, lane = tid & 31, wid = tid >> 5;
    int wy = wid >> 2, wx = wid & 3;                // 2 x 4 wave grid
    int e = blockIdx.z;
    const _Float16* A = p.A + (long)e * p.aE;
    const float*    W = p.W + (long)e * p.wE;
    long m0 = (long)blockIdx.y * 128;
    long n0 = (long)blockIdx.x * 128;

    v8f acc[4][2];
    #pragma unroll
    for (int i = 0; i < 4; ++i)
        #pragma unroll
        for (int j = 0; j < 2; ++j)
            #pragma unroll
            for (int r = 0; r < 8; ++r) acc[i][j][r] = 0.f;

    int ldRow = tid >> 1;
    int ldCol = (tid & 1) * 32;                     // element offset within row

    for (int k0 = 0; k0 < p.K; k0 += 64) {
        __syncthreads();
        // A tile: 128x64 f16 via async copy (4 x b128 per thread)
        {
            const _Float16* gp = A + (m0 + ldRow) * (long)p.K + k0 + ldCol;
            _Float16* lp = &sA[ldRow * 72 + ldCol];
            #pragma unroll
            for (int q = 0; q < 4; ++q) async_copy16(gp + q * 8, lp + q * 8);
        }
        // W tile: 128x64 f32 -> f16 through VGPRs
        {
            const float* wp = W + (n0 + ldRow) * (long)p.K + k0 + ldCol;
            union { _Float16 h[32]; uint4 u[4]; } cv;
            #pragma unroll
            for (int q = 0; q < 8; ++q) {
                float4 f = ((const float4*)wp)[q];
                cv.h[q*4+0] = (_Float16)f.x; cv.h[q*4+1] = (_Float16)f.y;
                cv.h[q*4+2] = (_Float16)f.z; cv.h[q*4+3] = (_Float16)f.w;
            }
            #pragma unroll
            for (int q = 0; q < 4; ++q)
                *(uint4*)&sB[ldRow * 72 + ldCol + q * 8] = cv.u[q];
            if (k0 + 64 < p.K) __builtin_prefetch(wp + 64, 0, 0);
        }
        wait_async0();
        __syncthreads();

        #pragma unroll
        for (int ks = 0; ks < 2; ++ks) {
            v16h a[4], b[2];
            #pragma unroll
            for (int i = 0; i < 4; ++i)
                a[i] = ld_frag(sA, wy * 64 + i * 16 + (lane & 15), 72, ks * 32, lane);
            #pragma unroll
            for (int j = 0; j < 2; ++j)
                b[j] = ld_frag(sB, wx * 32 + j * 16 + (lane & 15), 72, ks * 32, lane);
            #pragma unroll
            for (int i = 0; i < 4; ++i)
                #pragma unroll
                for (int j = 0; j < 2; ++j)
                    acc[i][j] = wmma16(a[i], b[j], acc[i][j]);
        }
    }

    // ---- epilogue
    const float* bias = p.bias ? (p.bias + (long)e * p.bE) : nullptr;
    #pragma unroll
    for (int i = 0; i < 4; ++i) {
        #pragma unroll
        for (int j = 0; j < 2; ++j) {
            #pragma unroll
            for (int r = 0; r < 8; ++r) {
                long m = m0 + wy * 64 + i * 16 + r + 8 * (lane >> 4);
                long n = n0 + wx * 32 + j * 16 + (lane & 15);
                float v = acc[i][j][r];
                if constexpr (MODE == GM_F16) {
                    (p.outH + (long)e * p.oE)[m * p.N + n] = (_Float16)v;
                } else if constexpr (MODE == GM_RESID) {
                    p.outF[m * p.N + n] = p.resid[m * p.N + n] + v;
                } else if constexpr (MODE == GM_GELU_F16) {
                    float t = v + bias[n];
                    (p.outH + (long)e * p.oE)[m * p.N + n] = (_Float16)gelu_tanh(t);
                } else { // GM_SCATTER
                    float t = v + bias[n];
                    const int*   si = p.selIdx + (long)e * p.selE;
                    const float* sw = p.selW   + (long)e * p.selE;
                    long tk = si[m];
                    atomicAdd(&p.scatter[tk * p.N + n], sw[m] * t);
                }
            }
        }
    }
}

// ---------------------------------------------------------------- QK-norm
// One wave per (token, head). Writes q16/k16 as (b,h,n,hd) and v transposed
// as (b,h,hd,n) so the P@V GEMM can use the weight-style B fragment loader.
__global__ __launch_bounds__(256) void qknorm_kernel(const _Float16* __restrict__ qkv,
                                                     const float* __restrict__ qg,
                                                     const float* __restrict__ qb,
                                                     const float* __restrict__ kg,
                                                     const float* __restrict__ kb,
                                                     _Float16* __restrict__ q16,
                                                     _Float16* __restrict__ k16,
                                                     _Float16* __restrict__ vT16) {
    int lane = threadIdx.x & 31, wid = threadIdx.x >> 5;
    long gw = (long)blockIdx.x * 8 + wid;           // 0..65535
    long tk = gw >> 4; int h = (int)(gw & 15);
    int b = (int)(tk >> 11); int n = (int)(tk & 2047);
    int bh = b * HH + h;
    const _Float16* base = qkv + tk * (3 * HH * HD_) + (long)h * HD_;

    float q0 = (float)base[lane],        q1 = (float)base[lane + 32];
    float k0 = (float)base[1024 + lane], k1 = (float)base[1024 + lane + 32];
    float v0 = (float)base[2048 + lane], v1 = (float)base[2048 + lane + 32];

    float qm = wave_sum(q0 + q1) * (1.0f / HD_);
    float qv = wave_sum(q0*q0 + q1*q1) * (1.0f / HD_) - qm * qm;
    float qr = rsqrtf(qv + 1e-5f);
    float km = wave_sum(k0 + k1) * (1.0f / HD_);
    float kv = wave_sum(k0*k0 + k1*k1) * (1.0f / HD_) - km * km;
    float kr = rsqrtf(kv + 1e-5f);

    long qBase = ((long)bh * NN + n) * HD_;
    q16[qBase + lane]      = (_Float16)((q0 - qm) * qr * qg[lane]      + qb[lane]);
    q16[qBase + lane + 32] = (_Float16)((q1 - qm) * qr * qg[lane + 32] + qb[lane + 32]);
    k16[qBase + lane]      = (_Float16)((k0 - km) * kr * kg[lane]      + kb[lane]);
    k16[qBase + lane + 32] = (_Float16)((k1 - km) * kr * kg[lane + 32] + kb[lane + 32]);
    vT16[((long)bh * HD_ + lane)      * NN + n] = (_Float16)v0;
    vT16[((long)bh * HD_ + lane + 32) * NN + n] = (_Float16)v1;
}

// ---------------------------------------------------------------- flash attn
// One WG per (64 query rows, b*h). Online softmax over 64-key blocks.
// Q/K/V^T tiles copied with CDNA5 async-to-LDS.
__global__ __launch_bounds__(256) void attn_kernel(const _Float16* __restrict__ q16,
                                                   const _Float16* __restrict__ k16,
                                                   const _Float16* __restrict__ vT16,
                                                   _Float16* __restrict__ o16) {
    __shared__ _Float16 Qs[64 * 72];
    __shared__ _Float16 Ks[64 * 72];
    __shared__ _Float16 Vt[64 * 72];
    __shared__ float    Sf[64 * 68];
    __shared__ _Float16 Ph[64 * 72];
    __shared__ float rowM[64], rowS[64], rowA[64];

    int tid = threadIdx.x, lane = tid & 31, wid = tid >> 5;
    int wy = wid >> 1, wx = wid & 1;                // 4 x 2 wave grid
    int bh = blockIdx.y;
    long q0 = (long)blockIdx.x * 64;
    const _Float16* qBase = q16 + ((long)bh * NN + q0) * HD_;
    const _Float16* kBase = k16 + (long)bh * NN * HD_;
    const _Float16* vBase = vT16 + (long)bh * HD_ * NN;

    int tr = tid >> 2, tcb = (tid & 3) * 16;
    {   // load Q (64 x 64) async
        const _Float16* src = qBase + (long)tr * HD_ + tcb;
        _Float16* dst = &Qs[tr * 72 + tcb];
        async_copy16(src, dst);
        async_copy16(src + 8, dst + 8);
    }
    if (tid < 64) { rowM[tid] = -1e30f; rowS[tid] = 0.f; }

    v8f oacc[2];
    #pragma unroll
    for (int j = 0; j < 2; ++j)
        #pragma unroll
        for (int r = 0; r < 8; ++r) oacc[j][r] = 0.f;

    for (int kb = 0; kb < NN / 64; ++kb) {
        __syncthreads();
        {   // K block (64 keys x 64) and V^T block (64 d x 64 keys) async
            const _Float16* ks = kBase + ((long)kb * 64 + tr) * HD_ + tcb;
            _Float16* kd = &Ks[tr * 72 + tcb];
            async_copy16(ks, kd);
            async_copy16(ks + 8, kd + 8);
            const _Float16* vs = vBase + (long)tr * NN + kb * 64 + tcb;
            _Float16* vd = &Vt[tr * 72 + tcb];
            async_copy16(vs, vd);
            async_copy16(vs + 8, vd + 8);
        }
        wait_async0();
        __syncthreads();

        // S = Q @ K^T (64x64), K-dim 64 -> two wmma steps
        v8f sacc[2];
        #pragma unroll
        for (int j = 0; j < 2; ++j)
            #pragma unroll
            for (int r = 0; r < 8; ++r) sacc[j][r] = 0.f;
        #pragma unroll
        for (int ks = 0; ks < 2; ++ks) {
            v16h a = ld_frag(Qs, wy * 16 + (lane & 15), 72, ks * 32, lane);
            #pragma unroll
            for (int j = 0; j < 2; ++j) {
                v16h b = ld_frag(Ks, wx * 32 + j * 16 + (lane & 15), 72, ks * 32, lane);
                sacc[j] = wmma16(a, b, sacc[j]);
            }
        }
        #pragma unroll
        for (int j = 0; j < 2; ++j)
            #pragma unroll
            for (int r = 0; r < 8; ++r) {
                int m = wy * 16 + r + 8 * (lane >> 4);
                int n = wx * 32 + j * 16 + (lane & 15);
                Sf[m * 68 + n] = sacc[j][r] * 0.125f;   // 1/sqrt(64)
            }
        __syncthreads();

        if (tid < 64) {                                 // online softmax, row tid
            float mOld = rowM[tid], mx = mOld;
            const float* sr = &Sf[tid * 68];
            #pragma unroll 8
            for (int j = 0; j < 64; ++j) mx = fmaxf(mx, sr[j]);
            float al = expf(mOld - mx);
            float sum = rowS[tid] * al;
            #pragma unroll 8
            for (int j = 0; j < 64; ++j) {
                float pj = expf(sr[j] - mx);
                Ph[tid * 72 + j] = (_Float16)pj;
                sum += pj;
            }
            rowM[tid] = mx; rowS[tid] = sum; rowA[tid] = al;
        }
        __syncthreads();

        // rescale O then O += P @ V
        #pragma unroll
        for (int j = 0; j < 2; ++j)
            #pragma unroll
            for (int r = 0; r < 8; ++r) {
                int m = wy * 16 + r + 8 * (lane >> 4);
                oacc[j][r] *= rowA[m];
            }
        #pragma unroll
        for (int ks = 0; ks < 2; ++ks) {
            v16h a = ld_frag(Ph, wy * 16 + (lane & 15), 72, ks * 32, lane);
            #pragma unroll
            for (int j = 0; j < 2; ++j) {
                v16h b = ld_frag(Vt, wx * 32 + j * 16 + (lane & 15), 72, ks * 32, lane);
                oacc[j] = wmma16(a, b, oacc[j]);
            }
        }
    }

    int b = bh >> 4, h = bh & 15;
    #pragma unroll
    for (int j = 0; j < 2; ++j)
        #pragma unroll
        for (int r = 0; r < 8; ++r) {
            int m = wy * 16 + r + 8 * (lane >> 4);
            int d = wx * 32 + j * 16 + (lane & 15);
            float inv = 1.0f / rowS[m];
            long tk = (long)b * NN + q0 + m;
            o16[tk * (HH * HD_) + h * HD_ + d] = (_Float16)(oacc[j][r] * inv);
        }
}

// ---------------------------------------------------------------- MoE gating
__global__ __launch_bounds__(256) void gate_kernel(const float* __restrict__ tok,
                                                   const float* __restrict__ Wg,
                                                   float* __restrict__ gate) {
    int lane = threadIdx.x & 31, wid = threadIdx.x >> 5;
    long tk = (long)blockIdx.x * 8 + wid;
    const float* xr = tok + tk * DD_;
    float acc[EE];
    #pragma unroll
    for (int e = 0; e < EE; ++e) acc[e] = 0.f;
    for (int k = lane; k < DD_; k += 32) {
        float xv = xr[k];
        #pragma unroll
        for (int e = 0; e < EE; ++e) acc[e] += xv * Wg[e * DD_ + k];
    }
    #pragma unroll
    for (int e = 0; e < EE; ++e) acc[e] = wave_sum(acc[e]);
    if (lane == 0) {
        float m = -1e30f;
        #pragma unroll
        for (int e = 0; e < EE; ++e) m = fmaxf(m, acc[e]);
        float s = 0.f;
        #pragma unroll
        for (int e = 0; e < EE; ++e) { acc[e] = expf(acc[e] - m); s += acc[e]; }
        float inv = 1.0f / s;
        #pragma unroll
        for (int e = 0; e < EE; ++e) gate[tk * EE + e] = acc[e] * inv;
    }
}

// per-expert top-KCAP via binary search on threshold + compaction
__global__ __launch_bounds__(256) void topk_kernel(const float* __restrict__ gate,
                                                   int* __restrict__ selIdx,
                                                   float* __restrict__ selW) {
    __shared__ int s_cnt, s_pos;
    int e = blockIdx.x, tid = threadIdx.x;
    for (int i = tid; i < KCAP; i += 256) {
        selIdx[e * KCAP + i] = 0; selW[e * KCAP + i] = 0.f;
    }
    float lo = 0.f, hi = 1.f;
    for (int it = 0; it < 24; ++it) {
        float mid = 0.5f * (lo + hi);
        if (tid == 0) s_cnt = 0;
        __syncthreads();
        int c = 0;
        for (int t = tid; t < TOK; t += 256) c += (gate[(long)t * EE + e] > mid);
        atomicAdd(&s_cnt, c);
        __syncthreads();
        int total = s_cnt;
        __syncthreads();
        if (total > KCAP) lo = mid; else hi = mid;
    }
    if (tid == 0) s_pos = 0;
    __syncthreads();
    for (int t = tid; t < TOK; t += 256) {            // definite members
        float gv = gate[(long)t * EE + e];
        if (gv > hi) {
            int p = atomicAdd(&s_pos, 1);
            if (p < KCAP) { selIdx[e * KCAP + p] = t; selW[e * KCAP + p] = gv; }
        }
    }
    __syncthreads();
    for (int t = tid; t < TOK; t += 256) {            // boundary fill
        float gv = gate[(long)t * EE + e];
        if (gv <= hi && gv > lo) {
            int p = atomicAdd(&s_pos, 1);
            if (p < KCAP) { selIdx[e * KCAP + p] = t; selW[e * KCAP + p] = gv; }
        }
    }
}

// ---------------------------------------------------------------- launcher
extern "C" void kernel_launch(void* const* d_in, const int* in_sizes, int n_in,
                              void* d_out, int out_size, void* d_ws, size_t ws_size,
                              hipStream_t stream) {
    (void)in_sizes; (void)n_in; (void)out_size; (void)ws_size;
    const float* x     = (const float*)d_in[0];
    const float* t     = (const float*)d_in[1];
    const float* W_ada = (const float*)d_in[2];
    const float* b_ada = (const float*)d_in[3];
    const float* ln1_g = (const float*)d_in[4];
    const float* ln1_b = (const float*)d_in[5];
    const float* Wqkv  = (const float*)d_in[6];
    const float* qn_g  = (const float*)d_in[7];
    const float* qn_b  = (const float*)d_in[8];
    const float* kn_g  = (const float*)d_in[9];
    const float* kn_b  = (const float*)d_in[10];
    const float* Wout  = (const float*)d_in[11];
    const float* mlp_g = (const float*)d_in[12];
    const float* mlp_b = (const float*)d_in[13];
    const float* Wgate = (const float*)d_in[14];
    const float* fc1s  = (const float*)d_in[15];
    const float* b1s   = (const float*)d_in[16];
    const float* fc2s  = (const float*)d_in[17];
    const float* b2s   = (const float*)d_in[18];
    float* out = (float*)d_out;

    char* ws = (char*)d_ws;
    size_t off = 0;
    auto carve = [&](size_t bytes) -> char* {
        char* p = ws + off;
        off = (off + bytes + 255) & ~(size_t)255;
        return p;
    };
    float*    shift = (float*)   carve((size_t)BB * DD_ * 4);
    _Float16* h16   = (_Float16*)carve((size_t)TOK * DD_ * 2);
    _Float16* qkv16 = (_Float16*)carve((size_t)TOK * 3 * DD_ * 2);
    _Float16* q16   = (_Float16*)carve((size_t)BB * HH * NN * HD_ * 2);
    _Float16* k16   = (_Float16*)carve((size_t)BB * HH * NN * HD_ * 2);
    _Float16* vT16  = (_Float16*)carve((size_t)BB * HH * HD_ * NN * 2);
    _Float16* o16   = (_Float16*)carve((size_t)TOK * DD_ * 2);
    float*    gate  = (float*)   carve((size_t)TOK * EE * 4);
    int*      selI  = (int*)     carve((size_t)EE * KCAP * 4);
    float*    selWv = (float*)   carve((size_t)EE * KCAP * 4);
    _Float16* g16   = (_Float16*)carve((size_t)TOK * DD_ * 2);
    _Float16* y16   = (_Float16*)carve((size_t)TOK * DFF * 2);

    // 1) AdaLN shift
    shift_kernel<<<8, 256, 0, stream>>>(t, W_ada, b_ada, shift);
    // 2) h = LN(x)*g + b + shift  (f16)
    ln_kernel<<<TOK, 256, 0, stream>>>(x, nullptr, shift, ln1_g, ln1_b, h16);
    // 3) qkv = h @ Wqkv^T
    {
        GemmArgs p{};
        p.A = h16; p.W = Wqkv; p.outH = qkv16;
        p.M = TOK; p.N = 3 * DD_; p.K = DD_;
        gemm_kernel<GM_F16><<<dim3((3 * DD_) / 128, TOK / 128, 1), 256, 0, stream>>>(p);
    }
    // 4) QK-norm + head-major repack (v transposed)
    qknorm_kernel<<<(TOK * HH) / 8, 256, 0, stream>>>(qkv16, qn_g, qn_b, kn_g, kn_b,
                                                      q16, k16, vT16);
    // 5) flash attention
    attn_kernel<<<dim3(NN / 64, BB * HH), 256, 0, stream>>>(q16, k16, vT16, o16);
    // 6) out = x + o @ Wout^T   (writes d_out == tok)
    {
        GemmArgs p{};
        p.A = o16; p.W = Wout; p.outF = out; p.resid = x;
        p.M = TOK; p.N = DD_; p.K = DD_;
        gemm_kernel<GM_RESID><<<dim3(DD_ / 128, TOK / 128, 1), 256, 0, stream>>>(p);
    }
    // 7) gating + top-k dispatch
    gate_kernel<<<TOK / 8, 256, 0, stream>>>(out, Wgate, gate);
    topk_kernel<<<EE, 256, 0, stream>>>(gate, selI, selWv);
    // 8) gather + LN dispatched tokens
    ln_kernel<<<EE * KCAP, 256, 0, stream>>>(out, selI, nullptr, mlp_g, mlp_b, g16);
    // 9) fc1 + gelu per expert
    {
        GemmArgs p{};
        p.A = g16; p.aE = (long)KCAP * DD_;
        p.W = fc1s; p.wE = (long)DFF * DD_;
        p.bias = b1s; p.bE = DFF;
        p.outH = y16; p.oE = (long)KCAP * DFF;
        p.M = KCAP; p.N = DFF; p.K = DD_;
        gemm_kernel<GM_GELU_F16><<<dim3(DFF / 128, KCAP / 128, EE), 256, 0, stream>>>(p);
    }
    // 10) fc2 + weighted scatter-add into tok (d_out)
    {
        GemmArgs p{};
        p.A = y16; p.aE = (long)KCAP * DFF;
        p.W = fc2s; p.wE = (long)DD_ * DFF;
        p.bias = b2s; p.bE = DD_;
        p.selIdx = selI; p.selW = selWv; p.selE = KCAP;
        p.scatter = out;
        p.M = KCAP; p.N = DD_; p.K = DFF;
        gemm_kernel<GM_SCATTER><<<dim3(DD_ / 128, KCAP / 128, EE), 256, 0, stream>>>(p);
    }
}
